// Attention_79259326480724
// MI455X (gfx1250) — matrix operand order repaired and checked by reading, not compiled, not run on hardware
//
#include <hip/hip_runtime.h>
#include <hip/hip_bf16.h>
#include <stdint.h>

// ---------------------------------------------------------------------------
// Channel attention (XCiT-style) for MI455X / gfx1250, bf16 WMMA pipeline.
// B=8, N=4096, C=512, H=8, DH=64
// Pipeline:
//   k_conv : x/w_qkv/w_proj -> bf16
//   k_qkv  : qkv GEMM (double-buffered async LDS staging), transposed bf16 out
//   k_attn1: K-split partial S = q@k^T + partial sumsq   (256 blocks)
//   k_attn2: reduce partials, L2-norm scale, softmax -> P (bf16)
//   k_attn3: N-split out = P@v, token-major bf16 y        (256 blocks)
//   k_proj : projection GEMM (double-buffered async), fp32 out
// ---------------------------------------------------------------------------

typedef __attribute__((ext_vector_type(16))) __bf16        v16bf;
typedef __attribute__((ext_vector_type(2)))  __bf16        v2bf;
typedef __attribute__((ext_vector_type(8)))  float         v8f;
typedef __attribute__((ext_vector_type(4)))  unsigned int  uint4v;
typedef __attribute__((ext_vector_type(2)))  unsigned int  uint2v;
typedef __attribute__((ext_vector_type(4)))  float         float4v;

#define DEV __device__ __forceinline__

#if defined(__has_builtin)
#if __has_builtin(__builtin_amdgcn_cvt_pk_bf16_f32)
#define HAVE_CVT_PK_BF16 1
#endif
#if __has_builtin(__builtin_amdgcn_s_wait_asynccnt)
#define HAVE_WAIT_ASYNC 1
#endif
#endif

DEV unsigned short f2bf(float f) {
  unsigned int u = __float_as_uint(f);
  u += 0x7FFFu + ((u >> 16) & 1u);          // round-to-nearest-even
  return (unsigned short)(u >> 16);
}

DEV unsigned int cvt2(float a, float b) {   // two fp32 -> packed bf16x2
#ifdef HAVE_CVT_PK_BF16
  v2bf r = __builtin_amdgcn_cvt_pk_bf16_f32(a, b);
  return __builtin_bit_cast(unsigned int, r);
#else
  return (unsigned int)f2bf(a) | ((unsigned int)f2bf(b) << 16);
#endif
}

DEV void wait_async0() {
#ifdef HAVE_WAIT_ASYNC
  __builtin_amdgcn_s_wait_asynccnt(0);
#else
  asm volatile("s_wait_asynccnt 0" ::: "memory");
#endif
}

// per-lane async copy of 16 bytes global -> LDS (tracked by ASYNCcnt)
DEV void async_ld_b128(unsigned int lds_off, const void* gaddr) {
  asm volatile("global_load_async_to_lds_b128 %0, %1, off"
               :: "v"(lds_off), "v"(gaddr) : "memory");
}

union V16 { uint4v u[2]; v16bf v; };

// A operand (16x32 bf16): lane<16 holds K {0..7,16..23}, lane>=16 {8..15,24..31}
DEV v16bf load_a_op(const unsigned short* row, int hi) {
  V16 t;
  t.u[0] = *(const uint4v*)(row + 8 * hi);
  t.u[1] = *(const uint4v*)(row + 16 + 8 * hi);
  return t.v;
}
// B operand (32x16 bf16): lane<16 holds K {0..7,16..23}, lane>=16 {8..15,24..31}
DEV v16bf load_b_op(const unsigned short* row, int hi) {
  V16 t;
  t.u[0] = *(const uint4v*)(row + 8 * hi);
  t.u[1] = *(const uint4v*)(row + 16 + 8 * hi);
  return t.v;
}

DEV v8f wmma_bf16(v16bf a, v16bf b, v8f c) {
  return __builtin_amdgcn_wmma_f32_16x16x32_bf16(false, a, false, b,
                                                 (short)0, c, false, false);
}

DEV uint4v pack8(const v8f& c) {
  uint4v r;
  r.x = cvt2(c[0], c[1]);
  r.y = cvt2(c[2], c[3]);
  r.z = cvt2(c[4], c[5]);
  r.w = cvt2(c[6], c[7]);
  return r;
}

DEV float sumsq8(uint4v v) {   // 8 packed bf16 -> sum of squares (fp32)
  float s = 0.f;
#pragma unroll
  for (int i = 0; i < 4; ++i) {
    unsigned int a = v[i];
    float lo = __uint_as_float(a << 16);
    float hf = __uint_as_float(a & 0xFFFF0000u);
    s += lo * lo + hf * hf;
  }
  return s;
}

DEV v8f vzero8() { return (v8f){0.f, 0.f, 0.f, 0.f, 0.f, 0.f, 0.f, 0.f}; }

// ---------------------------------------------------------------------------
// Kernel 0: convert x [B,N,C], w_qkv [1536,512], w_proj [512,512] -> bf16
// ---------------------------------------------------------------------------
__global__ __launch_bounds__(256)
void k_conv(const float* __restrict__ x, const float* __restrict__ wq,
            const float* __restrict__ wp,
            unsigned short* __restrict__ xb, unsigned short* __restrict__ wqb,
            unsigned short* __restrict__ wpb) {
  const int NX = (8 * 4096 * 512) / 4;    // 4,194,304 float4
  const int N1 = (1536 * 512) / 4;        //   196,608
  const int N2 = (512 * 512) / 4;         //    65,536
  int i = blockIdx.x * 256 + threadIdx.x;
  const float4v* src;
  unsigned short* dst;
  int j;
  if (i < NX)                { src = (const float4v*)x;  dst = xb;  j = i; }
  else if (i < NX + N1)      { src = (const float4v*)wq; dst = wqb; j = i - NX; }
  else if (i < NX + N1 + N2) { src = (const float4v*)wp; dst = wpb; j = i - NX - N1; }
  else return;
  float4v f = src[j];
  uint2v o;
  o.x = cvt2(f.x, f.y);
  o.y = cvt2(f.z, f.w);
  *(uint2v*)(dst + (size_t)j * 4) = o;
}

// ---------------------------------------------------------------------------
// Kernel 1: qkv = x @ w_qkv^T, stored TRANSPOSED bf16: qkvT[b*1536+j][n]
// 256 thr (8 waves), tile 128x128, BK=32, double-buffered async staging.
// ---------------------------------------------------------------------------
__global__ __launch_bounds__(256)
void k_qkv(const unsigned short* __restrict__ xb,
           const unsigned short* __restrict__ wqb,
           unsigned short* __restrict__ qkvT) {
  __shared__ __align__(16) unsigned short As[2][128 * 48];
  __shared__ __align__(16) unsigned short Bs[2][128 * 48];
  const unsigned int BUFSZ = 128 * 48 * 2;   // bytes per buffer

  const int tid = threadIdx.x;
  const int t0 = blockIdx.y * 128;          // token base
  const int j0 = blockIdx.x * 128;          // out-channel base
  const int w = tid >> 5, lane = tid & 31;
  const int wm = w >> 1, wn = w & 1;
  const int lr = lane & 15, hi = lane >> 4;

  const unsigned int as_base = (unsigned int)(uintptr_t)(void*)&As[0][0];
  const unsigned int bs_base = (unsigned int)(uintptr_t)(void*)&Bs[0][0];

  auto stage = [&](int buf, int kk) {
#pragma unroll
    for (int it = 0; it < 2; ++it) {
      int idx = tid + 256 * it;
      int row = idx >> 2, seg = idx & 3;
      unsigned int loff = (unsigned int)(row * 48 + seg * 8) * 2u + buf * BUFSZ;
      async_ld_b128(as_base + loff, xb + (size_t)(t0 + row) * 512 + kk + seg * 8);
      async_ld_b128(bs_base + loff, wqb + (size_t)(j0 + row) * 512 + kk + seg * 8);
    }
  };

  v8f acc[2][4];
#pragma unroll
  for (int i = 0; i < 2; ++i)
#pragma unroll
    for (int j = 0; j < 4; ++j) acc[i][j] = vzero8();

  stage(0, 0);
  wait_async0();
  __syncthreads();

  int buf = 0;
  for (int kk = 0; kk < 512; kk += 32) {
    if (kk + 32 < 512) stage(buf ^ 1, kk + 32);   // prefetch next chunk

    const unsigned short* Ab = &As[buf][0];
    const unsigned short* Bb = &Bs[buf][0];
    v16bf A0 = load_a_op(Ab + (wm * 32 + lr) * 48, hi);
    v16bf A1 = load_a_op(Ab + (wm * 32 + 16 + lr) * 48, hi);
    v16bf Bv[4];
#pragma unroll
    for (int j = 0; j < 4; ++j)
      Bv[j] = load_b_op(Bb + (wn * 64 + j * 16 + lr) * 48, hi);
#pragma unroll
    for (int j = 0; j < 4; ++j) {
      acc[0][j] = wmma_bf16(A0, Bv[j], acc[0][j]);
      acc[1][j] = wmma_bf16(A1, Bv[j], acc[1][j]);
    }
    wait_async0();
    __syncthreads();
    buf ^= 1;
  }

  // transposed bf16 store: each lane writes 8 consecutive tokens of one channel
  const int b = t0 >> 12;
  const int n0 = t0 & 4095;
#pragma unroll
  for (int i = 0; i < 2; ++i)
#pragma unroll
    for (int j = 0; j < 4; ++j) {
      int jg = j0 + wn * 64 + j * 16 + lr;
      int ng = n0 + wm * 32 + i * 16 + hi * 8;
      uint4v p = pack8(acc[i][j]);
      *(uint4v*)(qkvT + (((size_t)(b * 1536 + jg)) << 12) + ng) = p;
    }
}

// ---------------------------------------------------------------------------
// Kernel 2a: K-split partial S = q@k^T + partial sumsq.
// grid (4 K-splits, 64 pairs), 512 thr (16 waves, one 16x16 S tile each).
// ---------------------------------------------------------------------------
__global__ __launch_bounds__(512)
void k_attn1(const unsigned short* __restrict__ qkvT,
             float* __restrict__ Spart, float* __restrict__ Npart) {
  __shared__ __align__(16) unsigned short qs[2][64 * 80];
  __shared__ __align__(16) unsigned short ks[2][64 * 80];
  __shared__ float norms[128];                       // nq[64], nk[64]

  const int tid = threadIdx.x;
  const int ksp = blockIdx.x;                        // 0..3
  const int pair = blockIdx.y;                       // 0..63
  const int b = pair >> 3, h = pair & 7;
  const size_t qrow = (size_t)(b * 1536 + h * 64);
  const size_t krow = qrow + 512;
  const int nbase = ksp * 1024;

  const int w = tid >> 5, lane = tid & 31;
  const int md = w >> 2, me = w & 3;
  const int lr = lane & 15, hi = lane >> 4;
  const int srow = tid >> 3, sseg = tid & 7;

  if (tid < 128) norms[tid] = 0.f;

  // prefetch chunk 0 into VGPRs
  uint4v qv = *(const uint4v*)(qkvT + ((qrow + srow) << 12) + nbase + sseg * 8);
  uint4v kv = *(const uint4v*)(qkvT + ((krow + srow) << 12) + nbase + sseg * 8);
  __syncthreads();

  v8f acc = vzero8();
  float sq = 0.f, sk = 0.f;
  int buf = 0;

  for (int it = 0; it < 16; ++it) {
    *(uint4v*)(&qs[buf][srow * 80 + sseg * 8]) = qv;
    *(uint4v*)(&ks[buf][srow * 80 + sseg * 8]) = kv;
    sq += sumsq8(qv);
    sk += sumsq8(kv);
    if (it + 1 < 16) {   // prefetch next chunk while this one is consumed
      int nb = nbase + (it + 1) * 64;
      qv = *(const uint4v*)(qkvT + ((qrow + srow) << 12) + nb + sseg * 8);
      kv = *(const uint4v*)(qkvT + ((krow + srow) << 12) + nb + sseg * 8);
    }
    __syncthreads();
    const unsigned short* ar = &qs[buf][(md * 16 + lr) * 80];
    const unsigned short* br = &ks[buf][(me * 16 + lr) * 80];
    acc = wmma_bf16(load_a_op(ar, hi),      load_b_op(br, hi),      acc);
    acc = wmma_bf16(load_a_op(ar + 32, hi), load_b_op(br + 32, hi), acc);
    buf ^= 1;
  }

  atomicAdd(&norms[srow], sq);          // nq
  atomicAdd(&norms[64 + srow], sk);     // nk

  // store partial S tile directly from accumulators
  float* Sp = Spart + (size_t)(ksp * 64 + pair) * 4096;
#pragma unroll
  for (int v = 0; v < 8; ++v)
    Sp[(md * 16 + hi * 8 + v) * 64 + me * 16 + lr] = acc[v];

  __syncthreads();
  if (tid < 128) Npart[(size_t)(ksp * 64 + pair) * 128 + tid] = norms[tid];
}

// ---------------------------------------------------------------------------
// Kernel 2b: reduce partials, L2-norm scale + temperature, softmax -> P bf16
// grid (64 pairs), 64 threads (one row d each).
// ---------------------------------------------------------------------------
__global__ __launch_bounds__(64)
void k_attn2(const float* __restrict__ Spart, const float* __restrict__ Npart,
             const float* __restrict__ temperature,
             unsigned short* __restrict__ Pbuf) {
  __shared__ float invk[64];
  __shared__ float rowbuf[64 * 65];

  const int pair = blockIdx.x;
  const int d = threadIdx.x;
  const int h = pair & 7;

  float nqv = 0.f, nkv = 0.f;
#pragma unroll
  for (int s = 0; s < 4; ++s) {
    nqv += Npart[(size_t)(s * 64 + pair) * 128 + d];
    nkv += Npart[(size_t)(s * 64 + pair) * 128 + 64 + d];
  }
  float invq = 1.0f / fmaxf(sqrtf(nqv), 1e-12f);
  invk[d] = 1.0f / fmaxf(sqrtf(nkv), 1e-12f);
  __syncthreads();

  float* r = rowbuf + d * 65;
  float tq = invq * temperature[h];
  float m = -1e30f;
#pragma unroll 8
  for (int e = 0; e < 64; ++e) {
    float s = 0.f;
#pragma unroll
    for (int ks2 = 0; ks2 < 4; ++ks2)
      s += Spart[(size_t)(ks2 * 64 + pair) * 4096 + d * 64 + e];
    float a = s * tq * invk[e];
    r[e] = a;
    m = fmaxf(m, a);
  }
  float sum = 0.f;
#pragma unroll 8
  for (int e = 0; e < 64; ++e) {
    float p = __expf(r[e] - m);
    r[e] = p;
    sum += p;
  }
  float inv = 1.0f / sum;
#pragma unroll 8
  for (int e = 0; e < 64; ++e)
    Pbuf[(size_t)pair * 4096 + d * 64 + e] = f2bf(r[e] * inv);
}

// ---------------------------------------------------------------------------
// Kernel 2c: out = P @ v, N-split; token-major bf16 store into y[b][n][c].
// grid (4 N-splits, 64 pairs), 512 thr (16 waves: d-tile x n-slot).
// ---------------------------------------------------------------------------
__global__ __launch_bounds__(512)
void k_attn3(const unsigned short* __restrict__ qkvT,
             const unsigned short* __restrict__ Pbuf,
             unsigned short* __restrict__ yb) {
  __shared__ __align__(16) unsigned short P[64 * 80];
  __shared__ __align__(16) unsigned short vT[2][64 * 80];

  const int tid = threadIdx.x;
  const int ns4 = blockIdx.x;                        // 0..3
  const int pair = blockIdx.y;                       // 0..63
  const int b = pair >> 3, h = pair & 7;
  const size_t vrow = (size_t)(b * 1536 + 1024 + h * 64);
  const int nbase = ns4 * 1024;

  const int w = tid >> 5, lane = tid & 31;
  const int md = w >> 2, ns = w & 3;
  const int lr = lane & 15, hi = lane >> 4;
  const int srow = tid >> 3, sseg = tid & 7;

  // load P (bf16, row-major d x e) into LDS
  *(uint4v*)(&P[srow * 80 + sseg * 8]) =
      *(const uint4v*)(Pbuf + (size_t)pair * 4096 + srow * 64 + sseg * 8);

  // prefetch v chunk 0
  uint4v vv = *(const uint4v*)(qkvT + ((vrow + srow) << 12) + nbase + sseg * 8);
  __syncthreads();

  v16bf PA0 = load_a_op(&P[(md * 16 + lr) * 80], hi);
  v16bf PA1 = load_a_op(&P[(md * 16 + lr) * 80 + 32], hi);

  int buf = 0;
  for (int it = 0; it < 16; ++it) {
    const unsigned short* hv = (const unsigned short*)&vv;
#pragma unroll
    for (int i = 0; i < 8; ++i)
      vT[buf][(sseg * 8 + i) * 80 + srow] = hv[i];    // in-LDS transpose
    if (it + 1 < 16)
      vv = *(const uint4v*)(qkvT + ((vrow + srow) << 12) + nbase +
                            (it + 1) * 64 + sseg * 8);
    __syncthreads();

    const unsigned short* br = &vT[buf][(ns * 16 + lr) * 80];
    v8f o = vzero8();
    o = wmma_bf16(PA0, load_b_op(br, hi), o);
    o = wmma_bf16(PA1, load_b_op(br + 32, hi), o);

    int ng = nbase + it * 64 + ns * 16 + lr;
    size_t addr = (((size_t)(b * 4096 + ng)) << 9) + h * 64 + md * 16 + hi * 8;
    *(uint4v*)(yb + addr) = pack8(o);
    buf ^= 1;
  }
}

// ---------------------------------------------------------------------------
// Kernel 3: out = y @ w_proj^T (fp32 output, natural layout)
// 256 thr, tile 128x128, BK=32, double-buffered async staging.
// ---------------------------------------------------------------------------
__global__ __launch_bounds__(256)
void k_proj(const unsigned short* __restrict__ yb,
            const unsigned short* __restrict__ wpb,
            float* __restrict__ out) {
  __shared__ __align__(16) unsigned short As[2][128 * 48];
  __shared__ __align__(16) unsigned short Bs[2][128 * 48];
  const unsigned int BUFSZ = 128 * 48 * 2;

  const int tid = threadIdx.x;
  const int t0 = blockIdx.y * 128;
  const int j0 = blockIdx.x * 128;
  const int w = tid >> 5, lane = tid & 31;
  const int wm = w >> 1, wn = w & 1;
  const int lr = lane & 15, hi = lane >> 4;

  const unsigned int as_base = (unsigned int)(uintptr_t)(void*)&As[0][0];
  const unsigned int bs_base = (unsigned int)(uintptr_t)(void*)&Bs[0][0];

  auto stage = [&](int buf, int kk) {
#pragma unroll
    for (int it = 0; it < 2; ++it) {
      int idx = tid + 256 * it;
      int row = idx >> 2, seg = idx & 3;
      unsigned int loff = (unsigned int)(row * 48 + seg * 8) * 2u + buf * BUFSZ;
      async_ld_b128(as_base + loff, yb + (size_t)(t0 + row) * 512 + kk + seg * 8);
      async_ld_b128(bs_base + loff, wpb + (size_t)(j0 + row) * 512 + kk + seg * 8);
    }
  };

  v8f acc[2][4];
#pragma unroll
  for (int i = 0; i < 2; ++i)
#pragma unroll
    for (int j = 0; j < 4; ++j) acc[i][j] = vzero8();

  stage(0, 0);
  wait_async0();
  __syncthreads();

  int buf = 0;
  for (int kk = 0; kk < 512; kk += 32) {
    if (kk + 32 < 512) stage(buf ^ 1, kk + 32);

    const unsigned short* Ab = &As[buf][0];
    const unsigned short* Bb = &Bs[buf][0];
    v16bf A0 = load_a_op(Ab + (wm * 32 + lr) * 48, hi);
    v16bf A1 = load_a_op(Ab + (wm * 32 + 16 + lr) * 48, hi);
    v16bf Bv[4];
#pragma unroll
    for (int j = 0; j < 4; ++j)
      Bv[j] = load_b_op(Bb + (wn * 64 + j * 16 + lr) * 48, hi);
#pragma unroll
    for (int j = 0; j < 4; ++j) {
      acc[0][j] = wmma_bf16(A0, Bv[j], acc[0][j]);
      acc[1][j] = wmma_bf16(A1, Bv[j], acc[1][j]);
    }
    wait_async0();
    __syncthreads();
    buf ^= 1;
  }

#pragma unroll
  for (int i = 0; i < 2; ++i)
#pragma unroll
    for (int j = 0; j < 4; ++j) {
      int jg = j0 + wn * 64 + j * 16 + lr;
      int mg = t0 + wm * 32 + i * 16 + hi * 8;
#pragma unroll
      for (int v = 0; v < 8; ++v)
        out[(size_t)(mg + v) * 512 + jg] = acc[i][j][v];
    }
}

// ---------------------------------------------------------------------------
// Host launch
// ---------------------------------------------------------------------------
extern "C" void kernel_launch(void* const* d_in, const int* in_sizes, int n_in,
                              void* d_out, int out_size, void* d_ws, size_t ws_size,
                              hipStream_t stream) {
  const float* x    = (const float*)d_in[0];
  const float* wq   = (const float*)d_in[1];
  const float* wp   = (const float*)d_in[2];
  const float* temp = (const float*)d_in[3];
  float* out = (float*)d_out;

  char* ws = (char*)d_ws;
  unsigned short* wqb  = (unsigned short*)(ws);                 //   1,572,864 B
  unsigned short* wpb  = (unsigned short*)(ws + 1572864);       //     524,288 B
  unsigned short* xb   = (unsigned short*)(ws + 2097152);       //  33,554,432 B
  unsigned short* qkvT = (unsigned short*)(ws + 35651584);      // 100,663,296 B
  unsigned short* yb   = (unsigned short*)(ws + 136314880);     //  33,554,432 B
  float*          Spart= (float*)(ws + 169869312);              //   4,194,304 B
  float*          Npart= (float*)(ws + 174063616);              //     131,072 B
  unsigned short* Pbuf = (unsigned short*)(ws + 174194688);     //     524,288 B

  k_conv <<<dim3(17408),  dim3(256), 0, stream>>>(x, wq, wp, xb, wqb, wpb);
  k_qkv  <<<dim3(12, 256),dim3(256), 0, stream>>>(xb, wqb, qkvT);
  k_attn1<<<dim3(4, 64),  dim3(512), 0, stream>>>(qkvT, Spart, Npart);
  k_attn2<<<dim3(64),     dim3(64),  0, stream>>>(Spart, Npart, temp, Pbuf);
  k_attn3<<<dim3(4, 64),  dim3(512), 0, stream>>>(qkvT, Pbuf, yb);
  k_proj <<<dim3(4, 256), dim3(256), 0, stream>>>(yb, wpb, out);
}
